// RecursiveMamba2_v34_43250320671001
// MI455X (gfx1250) — compile-verified, matrix operand order checked by reading, not run on hardware
//
#include <hip/hip_runtime.h>
#include <hip/hip_bf16.h>
#include <math.h>

typedef __bf16 bf16;
typedef __attribute__((ext_vector_type(16))) __bf16 v16bf;
typedef __attribute__((ext_vector_type(8)))  float  v8f;
typedef __attribute__((ext_vector_type(4)))  int    v4i;

#define V_SZ     50288
#define D_MODEL  768
#define L_ALL    12
#define SPLIT_L  6
#define NSTATE   64
#define HDIM     64
#define DIN      1536
#define H_HEADS  24
#define KCONV    4
#define DPROJ    3224
#define CONVD    1664
#define RLORA    8
#define MTOK     1024       // B*T
#define TSEQ     512
#define BATCH    2
#define EPSV     1e-5f
#define N_LOOPS  3          // trace-time constant in the reference

// ------------------------------------------------ CDNA5 async global->LDS
#if __has_builtin(__builtin_amdgcn_global_load_async_to_lds_b128)
#define ASYNC_LDS 1
typedef __attribute__((address_space(1))) v4i* gv4i_p;
typedef __attribute__((address_space(3))) v4i* lv4i_p;
__device__ inline void async_ld16(const bf16* g, bf16* l) {
  __builtin_amdgcn_global_load_async_to_lds_b128(
      (gv4i_p)(v4i*)(void*)const_cast<bf16*>(g),
      (lv4i_p)(v4i*)(void*)l, 0, 0);
}
#endif

template <int CNT>
__device__ inline void wait_async() {
#ifdef ASYNC_LDS
 #if __has_builtin(__builtin_amdgcn_s_wait_asynccnt)
  __builtin_amdgcn_s_wait_asynccnt(CNT);
 #else
  asm volatile("s_wait_asynccnt %0" :: "i"(CNT) : "memory");
 #endif
#endif
}

// ---------------------------------------------------------------- reductions
__device__ inline float block_reduce_sum(float v) {
  #pragma unroll
  for (int o = 16; o > 0; o >>= 1) v += __shfl_xor(v, o, 32);
  __shared__ float red[8];
  int lane = threadIdx.x & 31, w = threadIdx.x >> 5;
  if (lane == 0) red[w] = v;
  __syncthreads();
  float t = (threadIdx.x < 8) ? red[threadIdx.x] : 0.f;
  if (w == 0) {
    #pragma unroll
    for (int o = 4; o > 0; o >>= 1) t += __shfl_xor(t, o, 32);
    if (lane == 0) red[0] = t;
  }
  __syncthreads();
  return red[0];
}

// ---------------------------------------------------------------- WMMA GEMM
// C[M,N] = A[M,K] * W[N,K]^T ;  A,W bf16 row-major (K contiguous), C f32.
// Block 256 thr = 8 waves (4 along M x 2 along N). Tile BM=128, BN=128, BK=32.
// Wave tile 32x64 = 2x4 WMMA fragments. Double-buffered LDS, async loads.
#define BM 128
#define BN 128
#define BK 32
#define LDP 40   // padded LDS row (bf16 elems): 80B stride -> conflict-free b128 reads

__global__ __launch_bounds__(256) void gemm_bf16_nt(
    const bf16* __restrict__ A, const bf16* __restrict__ W,
    float* __restrict__ C, int M, int N, int K)
{
  __shared__ bf16 sA[2][BM][LDP];
  __shared__ bf16 sB[2][BN][LDP];
  const int tid  = threadIdx.x;
  const int m0   = blockIdx.y * BM;
  const int n0   = blockIdx.x * BN;
  const int lane = tid & 31;
  const int wid  = tid >> 5;     // 0..7
  const int wm   = wid >> 1;     // 0..3 (M)
  const int wn   = wid & 1;      // 0..1 (N)
  const int lr   = lane & 15;
  const int hx   = lane >> 4;    // half: 0/1

  // per-thread tile-load coordinates (one 16B chunk of A and of W each)
  const int lrow = tid >> 1;
  const int lch  = (tid & 1) * 16;
  int gn = n0 + lrow; if (gn > N - 1) gn = N - 1;   // clamp: garbage cols >= N never stored

  v8f acc[2][4];
  #pragma unroll
  for (int i = 0; i < 2; ++i)
    #pragma unroll
    for (int j = 0; j < 4; ++j) acc[i][j] = (v8f){};

  auto load_tile = [&](int kk, int buf) {
    const bf16* ga = A + (size_t)(m0 + lrow) * K + kk + lch;
    const bf16* gb = W + (size_t)gn * K + kk + lch;
#ifdef ASYNC_LDS
    async_ld16(ga, &sA[buf][lrow][lch]);
    async_ld16(gb, &sB[buf][lrow][lch]);
#else
    uint4 va = *(const uint4*)ga;
    uint4 vb = *(const uint4*)gb;
    *(uint4*)&sA[buf][lrow][lch] = va;
    *(uint4*)&sB[buf][lrow][lch] = vb;
#endif
  };

  load_tile(0, 0);
  int cur = 0;
  for (int kk = 0; kk < K; kk += BK) {
    const bool has_next = (kk + BK) < K;
    if (has_next) load_tile(kk + BK, cur ^ 1);
    if (has_next) wait_async<2>(); else wait_async<0>();
    __syncthreads();

    // fragment loads per ISA 16-bit A/B layout: lane lr, half hx,
    // elems 0..7 -> K = 8*hx.. ; elems 8..15 -> K = 16+8*hx..
    v16bf af[2], bfr[4];
    #pragma unroll
    for (int i = 0; i < 2; ++i) {
      const bf16* r = &sA[cur][wm * 32 + i * 16 + lr][0];
      *((uint4*)&af[i])     = *(const uint4*)(r + 8 * hx);
      *((uint4*)&af[i] + 1) = *(const uint4*)(r + 16 + 8 * hx);
    }
    #pragma unroll
    for (int j = 0; j < 4; ++j) {
      const bf16* r = &sB[cur][wn * 64 + j * 16 + lr][0];
      *((uint4*)&bfr[j])     = *(const uint4*)(r + 8 * hx);
      *((uint4*)&bfr[j] + 1) = *(const uint4*)(r + 16 + 8 * hx);
    }
    #pragma unroll
    for (int i = 0; i < 2; ++i)
      #pragma unroll
      for (int j = 0; j < 4; ++j)
        acc[i][j] = __builtin_amdgcn_wmma_f32_16x16x32_bf16(
            false, af[i], false, bfr[j], (short)0, acc[i][j], false, false);
    __syncthreads();
    cur ^= 1;
  }

  // C/D layout: lane -> n = lr ; VGPR v -> m = v + 8*hx (within a 16-row frag)
  #pragma unroll
  for (int i = 0; i < 2; ++i) {
    const int mB = m0 + wm * 32 + i * 16 + hx * 8;
    #pragma unroll
    for (int j = 0; j < 4; ++j) {
      const int nc = n0 + wn * 64 + j * 16 + lr;
      if (nc < N) {
        #pragma unroll
        for (int v = 0; v < 8; ++v)
          C[(size_t)(mB + v) * N + nc] = acc[i][j][v];
      }
    }
  }
}

// ---------------------------------------------------------------- elementwise
__global__ __launch_bounds__(256) void embed_kernel(
    const int* __restrict__ ids, const float* __restrict__ emb, float* __restrict__ x)
{
  int row = blockIdx.x;                 // 0..1023
  int tok = ids[row];
  for (int d = threadIdx.x; d < D_MODEL; d += 256)
    x[(size_t)row * D_MODEL + d] = emb[(size_t)tok * D_MODEL + d];
}

__global__ __launch_bounds__(256) void f32_to_bf16_kernel(
    const float* __restrict__ in, bf16* __restrict__ out, int n)
{
  int i = blockIdx.x * 256 + threadIdx.x;
  if (i < n) out[i] = (bf16)in[i];
}

__global__ __launch_bounds__(256) void add_kernel(
    float* __restrict__ c, const float* __restrict__ a, const float* __restrict__ b, int n)
{
  int i = blockIdx.x * 256 + threadIdx.x;
  if (i < n) c[i] = a[i] + b[i];
}

// eff[n,k] = base[n,k] + 2 * sum_r B[n,r] * A[r,k]  -> bf16
__global__ __launch_bounds__(256) void lora_eff_kernel(
    const float* __restrict__ base, const float* __restrict__ Bm,
    const float* __restrict__ Am, bf16* __restrict__ out, int N, int K)
{
  int id = blockIdx.x * 256 + threadIdx.x;
  if (id >= N * K) return;
  int n = id / K, k = id - n * K;
  float acc = base[id];
  #pragma unroll
  for (int r = 0; r < RLORA; ++r)
    acc += 2.0f * Bm[n * RLORA + r] * Am[r * K + k];
  out[id] = (bf16)acc;
}

template <typename OUT>
__global__ __launch_bounds__(256) void rms_kernel(
    const float* __restrict__ x, const float* __restrict__ w,
    OUT* __restrict__ out, int Dd)
{
  int row = blockIdx.x;
  const float* xr = x + (size_t)row * Dd;
  float vals[8];
  float ss = 0.f;
  int cnt = 0;
  for (int j = threadIdx.x; j < Dd; j += 256) { float v = xr[j]; vals[cnt++] = v; ss += v * v; }
  ss = block_reduce_sum(ss);
  float scale = rsqrtf(ss / (float)Dd + EPSV);
  cnt = 0;
  for (int j = threadIdx.x; j < Dd; j += 256)
    out[(size_t)row * Dd + j] = (OUT)(vals[cnt++] * scale * w[j]);
}

// causal depthwise conv (width 4) over time + SiLU; input columns [1536,3200) of zx
__global__ __launch_bounds__(256) void conv_silu_kernel(
    const float* __restrict__ zx, const float* __restrict__ cw,
    const float* __restrict__ cb, float* __restrict__ xbc)
{
  int id = blockIdx.x * 256 + threadIdx.x;
  if (id >= MTOK * CONVD) return;
  int m = id / CONVD, c = id - m * CONVD;
  int b = m / TSEQ, t = m - b * TSEQ;
  float acc = cb[c];
  #pragma unroll
  for (int k = 0; k < KCONV; ++k) {
    int tt = t + k - (KCONV - 1);
    if (tt >= 0)
      acc += cw[c * KCONV + k] * zx[(size_t)(b * TSEQ + tt) * DPROJ + DIN + c];
  }
  xbc[id] = acc / (1.f + __expf(-acc)) ;
}

__global__ __launch_bounds__(256) void dt_softplus_kernel(
    const float* __restrict__ zx, const float* __restrict__ dtb, float* __restrict__ dt)
{
  int id = blockIdx.x * 256 + threadIdx.x;
  if (id >= MTOK * H_HEADS) return;
  int m = id / H_HEADS, h = id - m * H_HEADS;
  float v = zx[(size_t)m * DPROJ + DIN + CONVD + h] + dtb[h];
  dt[id] = (v > 20.f) ? v : log1pf(__expf(v));
}

// SSM scan: 48 blocks = (b,h); 256 thr: p = tid>>2 (0..63), ng = tid&3 (n chunk of 16)
__global__ __launch_bounds__(256) void ssm_scan_kernel(
    const float* __restrict__ xbc, const float* __restrict__ dt,
    const float* __restrict__ A_log, const float* __restrict__ Dsk,
    float* __restrict__ y)
{
  int bh = blockIdx.x;
  int b = bh / H_HEADS, h = bh - b * H_HEADS;
  float A  = -__expf(A_log[h]);
  float Dh = Dsk[h];
  int p  = threadIdx.x >> 2;
  int ng = threadIdx.x & 3;
  int n0 = ng * 16;
  float hreg[16];
  #pragma unroll
  for (int i = 0; i < 16; ++i) hreg[i] = 0.f;

  __shared__ float sx[64], sB[64], sC[64], sdt[1];
  for (int t = 0; t < TSEQ; ++t) {
    int m = b * TSEQ + t;
    size_t base = (size_t)m * CONVD;
    int tid = threadIdx.x;
    if (tid < 64)        sx[tid]       = xbc[base + h * HDIM + tid];
    else if (tid < 128)  sB[tid - 64]  = xbc[base + DIN + (tid - 64)];
    else if (tid < 192)  sC[tid - 128] = xbc[base + DIN + NSTATE + (tid - 128)];
    else if (tid == 192) sdt[0]        = dt[(size_t)m * H_HEADS + h];
    __syncthreads();

    float dtv  = sdt[0];
    float dA   = __expf(dtv * A);
    float coef = dtv * sx[p];
    float acc = 0.f;
    #pragma unroll
    for (int i = 0; i < 16; ++i) {
      hreg[i] = dA * hreg[i] + coef * sB[n0 + i];
      acc += hreg[i] * sC[n0 + i];
    }
    acc += __shfl_xor(acc, 1, 32);
    acc += __shfl_xor(acc, 2, 32);
    if (ng == 0) y[(size_t)m * DIN + h * HDIM + p] = acc + Dh * sx[p];
    __syncthreads();
  }
}

// v = y * silu(z);  out = rms(v) * gw  (bf16 for GEMM A)
__global__ __launch_bounds__(256) void gate_rms_kernel(
    const float* __restrict__ y, const float* __restrict__ zx,
    const float* __restrict__ gw, bf16* __restrict__ out)
{
  int row = blockIdx.x;
  float vals[6];
  float ss = 0.f;
  int cnt = 0;
  for (int j = threadIdx.x; j < DIN; j += 256) {
    float z = zx[(size_t)row * DPROJ + j];
    float v = y[(size_t)row * DIN + j] * (z / (1.f + __expf(-z)));
    vals[cnt++] = v;
    ss += v * v;
  }
  ss = block_reduce_sum(ss);
  float scale = rsqrtf(ss / (float)DIN + EPSV);
  cnt = 0;
  for (int j = threadIdx.x; j < DIN; j += 256)
    out[(size_t)row * DIN + j] = (bf16)(vals[cnt++] * scale * gw[j]);
}

// x = rope(x + gate*xp, loop_i)  in place (each thread owns one even/odd pair)
__global__ __launch_bounds__(256) void gate_rope_kernel(
    float* __restrict__ x, const float* __restrict__ xp,
    const float* __restrict__ gate, float li)
{
  int id = blockIdx.x * 256 + threadIdx.x;
  if (id >= MTOK * (D_MODEL / 2)) return;
  int m = id / (D_MODEL / 2), dp = id - m * (D_MODEL / 2);
  int d0 = 2 * dp;
  size_t o = (size_t)m * D_MODEL + d0;
  float a = x[o]     + gate[d0]     * xp[o];
  float b = x[o + 1] + gate[d0 + 1] * xp[o + 1];
  float f = li * __powf(10000.f, -(float)d0 / (float)D_MODEL);
  float c = __cosf(f), s = __sinf(f);
  x[o]     = a * c - b * s;
  x[o + 1] = b * c + a * s;
}

// ---------------------------------------------------------------- host side
struct Ws {
  bf16 *Wi, *Wo, *lm;
  float *x, *res, *xp, *tmp;
  bf16 *u_bf, *y_bf;
  float *zx, *xbc, *dtv, *y;
};

static inline dim3 gemm_grid(int M, int N) { return dim3((N + BN - 1) / BN, M / BM); }

static void launch_mixer(hipStream_t s, Ws& w,
                         const float* u_src, bool do_rms, const float* nw,
                         const bf16* Wi, const float* cw, const float* cb,
                         const float* dtb, const float* al, const float* dk,
                         const float* gw, const bf16* Wo, float* out)
{
  if (do_rms)
    rms_kernel<bf16><<<MTOK, 256, 0, s>>>(u_src, nw, w.u_bf, D_MODEL);
  else
    f32_to_bf16_kernel<<<(MTOK * D_MODEL + 255) / 256, 256, 0, s>>>(u_src, w.u_bf, MTOK * D_MODEL);

  gemm_bf16_nt<<<gemm_grid(MTOK, DPROJ), 256, 0, s>>>(w.u_bf, Wi, w.zx, MTOK, DPROJ, D_MODEL);
  conv_silu_kernel<<<(MTOK * CONVD + 255) / 256, 256, 0, s>>>(w.zx, cw, cb, w.xbc);
  dt_softplus_kernel<<<(MTOK * H_HEADS + 255) / 256, 256, 0, s>>>(w.zx, dtb, w.dtv);
  ssm_scan_kernel<<<BATCH * H_HEADS, 256, 0, s>>>(w.xbc, w.dtv, al, dk, w.y);
  gate_rms_kernel<<<MTOK, 256, 0, s>>>(w.y, w.zx, gw, w.y_bf);
  gemm_bf16_nt<<<gemm_grid(MTOK, D_MODEL), 256, 0, s>>>(w.y_bf, Wo, out, MTOK, D_MODEL, DIN);
}

extern "C" void kernel_launch(void* const* d_in, const int* in_sizes, int n_in,
                              void* d_out, int out_size, void* d_ws, size_t ws_size,
                              hipStream_t stream)
{
  (void)in_sizes; (void)n_in; (void)out_size; (void)ws_size;

  const int*   ids     = (const int*)  d_in[0];
  // d_in[1] = n_loops : trace-time constant (3) in the reference; not readable during capture
  const float* emb     = (const float*)d_in[2];
  const float* norm_w  = (const float*)d_in[3];
  const float* in_w    = (const float*)d_in[4];
  const float* conv_w  = (const float*)d_in[5];
  const float* conv_b  = (const float*)d_in[6];
  const float* dt_bias = (const float*)d_in[7];
  const float* A_log   = (const float*)d_in[8];
  const float* D_skip  = (const float*)d_in[9];
  const float* gnorm   = (const float*)d_in[10];
  const float* out_w   = (const float*)d_in[11];
  const float* lAi     = (const float*)d_in[12];   // (6, R, D)
  const float* lBi     = (const float*)d_in[13];   // (6, DPROJ, R)
  const float* lAo     = (const float*)d_in[14];   // (6, R, DIN)
  const float* lBo     = (const float*)d_in[15];   // (6, D, R)
  const float* c_iw    = (const float*)d_in[16];
  const float* c_cw    = (const float*)d_in[17];
  const float* c_cb    = (const float*)d_in[18];
  const float* c_dtb   = (const float*)d_in[19];
  const float* c_al    = (const float*)d_in[20];
  const float* c_dk    = (const float*)d_in[21];
  const float* c_gw    = (const float*)d_in[22];
  const float* c_ow    = (const float*)d_in[23];
  const float* loop_nw = (const float*)d_in[24];
  const float* gate    = (const float*)d_in[25];
  const float* norm_f  = (const float*)d_in[26];
  const float* lm_head = (const float*)d_in[27];

  // ---- workspace carve (bump allocator, 256B aligned) ----
  char* p = (char*)d_ws;
  auto alloc = [&](size_t bytes) -> void* {
    void* r = (void*)p; p += (bytes + 255) & ~(size_t)255; return r;
  };
  const size_t WI_STRIDE = (size_t)DPROJ * D_MODEL;   // per-slot elements
  const size_t WO_STRIDE = (size_t)D_MODEL * DIN;

  Ws w;
  w.Wi   = (bf16*) alloc(13 * WI_STRIDE * sizeof(bf16));   // slots 0..11 layers, 12 core
  w.Wo   = (bf16*) alloc(13 * WO_STRIDE * sizeof(bf16));
  w.lm   = (bf16*) alloc((size_t)V_SZ * D_MODEL * sizeof(bf16));
  w.x    = (float*)alloc((size_t)MTOK * D_MODEL * 4);
  w.res  = (float*)alloc((size_t)MTOK * D_MODEL * 4);
  w.xp   = (float*)alloc((size_t)MTOK * D_MODEL * 4);
  w.tmp  = (float*)alloc((size_t)MTOK * D_MODEL * 4);
  w.u_bf = (bf16*) alloc((size_t)MTOK * D_MODEL * sizeof(bf16));
  w.y_bf = (bf16*) alloc((size_t)MTOK * DIN * sizeof(bf16));
  w.zx   = (float*)alloc((size_t)MTOK * DPROJ * 4);
  w.xbc  = (float*)alloc((size_t)MTOK * CONVD * 4);
  w.dtv  = (float*)alloc((size_t)MTOK * H_HEADS * 4);
  w.y    = (float*)alloc((size_t)MTOK * DIN * 4);

  const int ND = MTOK * D_MODEL;

  // ---- weight prep: bf16 conversion + LoRA merge ----
  for (int l = 0; l < SPLIT_L; ++l) {
    f32_to_bf16_kernel<<<(int)((WI_STRIDE + 255) / 256), 256, 0, stream>>>(
        in_w + l * WI_STRIDE, w.Wi + l * WI_STRIDE, (int)WI_STRIDE);
    f32_to_bf16_kernel<<<(int)((WO_STRIDE + 255) / 256), 256, 0, stream>>>(
        out_w + l * WO_STRIDE, w.Wo + l * WO_STRIDE, (int)WO_STRIDE);
  }
  for (int l = 0; l < L_ALL - SPLIT_L; ++l) {
    int g = SPLIT_L + l;
    lora_eff_kernel<<<(int)((WI_STRIDE + 255) / 256), 256, 0, stream>>>(
        in_w + g * WI_STRIDE, lBi + (size_t)l * DPROJ * RLORA,
        lAi + (size_t)l * RLORA * D_MODEL, w.Wi + g * WI_STRIDE, DPROJ, D_MODEL);
    lora_eff_kernel<<<(int)((WO_STRIDE + 255) / 256), 256, 0, stream>>>(
        out_w + g * WO_STRIDE, lBo + (size_t)l * D_MODEL * RLORA,
        lAo + (size_t)l * RLORA * DIN, w.Wo + g * WO_STRIDE, D_MODEL, DIN);
  }
  f32_to_bf16_kernel<<<(int)((WI_STRIDE + 255) / 256), 256, 0, stream>>>(
      c_iw, w.Wi + 12 * WI_STRIDE, (int)WI_STRIDE);
  f32_to_bf16_kernel<<<(int)((WO_STRIDE + 255) / 256), 256, 0, stream>>>(
      c_ow, w.Wo + 12 * WO_STRIDE, (int)WO_STRIDE);
  f32_to_bf16_kernel<<<(int)(((size_t)V_SZ * D_MODEL + 255) / 256), 256, 0, stream>>>(
      lm_head, w.lm, V_SZ * D_MODEL);

  // ---- forward ----
  embed_kernel<<<MTOK, 256, 0, stream>>>(ids, emb, w.x);
  (void)hipMemsetAsync(w.res, 0, (size_t)ND * 4, stream);

  // base + lora stacks (layers 0..11, effective weights already merged)
  for (int l = 0; l < L_ALL; ++l) {
    add_kernel<<<(ND + 255) / 256, 256, 0, stream>>>(w.res, w.res, w.x, ND);
    launch_mixer(stream, w, w.res, true, norm_w + l * D_MODEL,
                 w.Wi + l * WI_STRIDE, conv_w + (size_t)l * CONVD * KCONV,
                 conv_b + (size_t)l * CONVD, dt_bias + l * H_HEADS,
                 A_log + l * H_HEADS, D_skip + l * H_HEADS,
                 gnorm + (size_t)l * DIN, w.Wo + l * WO_STRIDE, w.x);
  }

  (void)hipMemcpyAsync(w.xp, w.x, (size_t)ND * 4, hipMemcpyDeviceToDevice, stream);

  for (int i = 0; i < N_LOOPS; ++i) {
    gate_rope_kernel<<<(MTOK * (D_MODEL / 2) + 255) / 256, 256, 0, stream>>>(
        w.x, w.xp, gate, (float)i);
    for (int l = SPLIT_L; l < L_ALL; ++l) {
      add_kernel<<<(ND + 255) / 256, 256, 0, stream>>>(w.res, w.res, w.x, ND);
      launch_mixer(stream, w, w.res, true, norm_w + l * D_MODEL,
                   w.Wi + l * WI_STRIDE, conv_w + (size_t)l * CONVD * KCONV,
                   conv_b + (size_t)l * CONVD, dt_bias + l * H_HEADS,
                   A_log + l * H_HEADS, D_skip + l * H_HEADS,
                   gnorm + (size_t)l * DIN, w.Wo + l * WO_STRIDE, w.x);
    }
    // core mixer on x directly (no pre-RMS), residual add, loop norm
    launch_mixer(stream, w, w.x, false, nullptr,
                 w.Wi + 12 * WI_STRIDE, c_cw, c_cb, c_dtb, c_al, c_dk, c_gw,
                 w.Wo + 12 * WO_STRIDE, w.tmp);
    add_kernel<<<(ND + 255) / 256, 256, 0, stream>>>(w.x, w.x, w.tmp, ND);
    rms_kernel<float><<<MTOK, 256, 0, stream>>>(w.x, loop_nw, w.tmp, D_MODEL);
    (void)hipMemcpyAsync(w.x, w.tmp, (size_t)ND * 4, hipMemcpyDeviceToDevice, stream);
  }

  // final: h = rms(x + res, norm_f) ; logits = h @ lm_head^T
  add_kernel<<<(ND + 255) / 256, 256, 0, stream>>>(w.tmp, w.x, w.res, ND);
  rms_kernel<bf16><<<MTOK, 256, 0, stream>>>(w.tmp, norm_f, w.u_bf, D_MODEL);
  gemm_bf16_nt<<<gemm_grid(MTOK, V_SZ), 256, 0, stream>>>(
      w.u_bf, w.lm, (float*)d_out, MTOK, V_SZ, D_MODEL);
}
